// CausalSelfAttention_42795054137680
// MI455X (gfx1250) — matrix operand order repaired and checked by reading, not compiled, not run on hardware
//
#include <hip/hip_runtime.h>

// ---- shapes (fixed by the reference) ----
#define B_  2
#define T_  2048
#define C_  1024
#define H_  16
#define HD  64
#define N3C (3 * C_)

typedef _Float16 h2  __attribute__((ext_vector_type(2)));
typedef _Float16 h8  __attribute__((ext_vector_type(8)));
typedef _Float16 h16 __attribute__((ext_vector_type(16)));
typedef float    f8  __attribute__((ext_vector_type(8)));
typedef float    f4  __attribute__((ext_vector_type(4)));
typedef unsigned int u32x4 __attribute__((ext_vector_type(4)));
typedef int          i32x4 __attribute__((ext_vector_type(4)));
typedef int          i32x8 __attribute__((ext_vector_type(8)));

#ifndef __has_builtin
#define __has_builtin(x) 0
#endif
#if __has_builtin(__builtin_amdgcn_tensor_load_to_lds) && \
    __has_builtin(__builtin_amdgcn_s_wait_tensorcnt)
#define USE_TDM 1
#else
#define USE_TDM 0
#endif
#if __has_builtin(__builtin_amdgcn_global_load_async_to_lds_b128)
#define USE_ASYNC 1
#else
#define USE_ASYNC 0
#endif

__device__ __forceinline__ h16 mk16(h8 lo, h8 hi) {
  h16 r;
#pragma unroll
  for (int i = 0; i < 8; ++i) { r[i] = lo[i]; r[i + 8] = hi[i]; }
  return r;
}
__device__ __forceinline__ f8 fzero8() {
  f8 z = {0.f, 0.f, 0.f, 0.f, 0.f, 0.f, 0.f, 0.f};
  return z;
}
__device__ __forceinline__ void wait_async0() {
#if __has_builtin(__builtin_amdgcn_s_wait_asynccnt)
  __builtin_amdgcn_s_wait_asynccnt(0);
#else
  asm volatile("s_wait_asynccnt 0x0" ::: "memory");
#endif
}
#define WMMA(a, b, c) \
  __builtin_amdgcn_wmma_f32_16x16x32_f16(false, (a), false, (b), (short)0, (c), false, false)

// =====================================================================
// Kernel 1: qkv = x @ W_qkv + b_qkv  ->  f16 Q/K/V in [B,H,T,HD] layout
// M=4096, N=3072, K=1024. 256 thr (8 waves, 4x2), tile 128x128, BK=32.
// =====================================================================
__global__ void __launch_bounds__(256) qkv_gemm_kernel(
    const float* __restrict__ x, const float* __restrict__ Wqkv,
    const float* __restrict__ bqkv,
    _Float16* __restrict__ Qh, _Float16* __restrict__ Kh, _Float16* __restrict__ Vh) {
  __shared__ _Float16 As[128 * 40];  // [m][k], stride 40 halves
  __shared__ _Float16 Bs[128 * 40];  // [n][k] (transposed W), stride 40
  const int tid = threadIdx.x;
  const int lane = tid & 31, wv = tid >> 5;
  const int wr = wv >> 1, wc = wv & 1;
  const int r = lane & 15, kb = lane >> 4;
  const int m0 = blockIdx.y * 128, n0 = blockIdx.x * 128;

  f8 acc[2][4];
#pragma unroll
  for (int mi = 0; mi < 2; ++mi)
#pragma unroll
    for (int ni = 0; ni < 4; ++ni) acc[mi][ni] = fzero8();

  for (int k0 = 0; k0 < C_; k0 += 32) {
    {  // stage A: x[128 x 32] -> f16
      const int m = tid >> 1, kk = (tid & 1) * 16;
      const float* src = x + (size_t)(m0 + m) * C_ + k0 + kk;
      if (k0 + 32 < C_) __builtin_prefetch(src + 32, 0, 1);
      f4 a0 = *(const f4*)(src), a1 = *(const f4*)(src + 4);
      f4 a2 = *(const f4*)(src + 8), a3 = *(const f4*)(src + 12);
      h8 h0, h1;
#pragma unroll
      for (int i = 0; i < 4; ++i) {
        h0[i] = (_Float16)a0[i]; h0[4 + i] = (_Float16)a1[i];
        h1[i] = (_Float16)a2[i]; h1[4 + i] = (_Float16)a3[i];
      }
      *(h8*)&As[m * 40 + kk] = h0;
      *(h8*)&As[m * 40 + kk + 8] = h1;
    }
    {  // stage B transposed: W[32 x 128] -> Bs[n][k], packed-pair b32 stores
      const int kkp = (tid >> 4) * 2;   // 0,2,...,30
      const int nb = (tid & 15) * 8;    // 0..120
      const float* s0 = Wqkv + (size_t)(k0 + kkp) * N3C + n0 + nb;
      const float* s1 = s0 + N3C;
      if (k0 + 32 < C_) __builtin_prefetch(s0 + (size_t)32 * N3C, 0, 1);
      f4 a0 = *(const f4*)(s0), a1 = *(const f4*)(s0 + 4);
      f4 b0 = *(const f4*)(s1), b1 = *(const f4*)(s1 + 4);
#pragma unroll
      for (int i = 0; i < 4; ++i) {
        h2 t0 = {(_Float16)a0[i], (_Float16)b0[i]};
        h2 t1 = {(_Float16)a1[i], (_Float16)b1[i]};
        *(h2*)&Bs[(nb + i) * 40 + kkp] = t0;
        *(h2*)&Bs[(nb + 4 + i) * 40 + kkp] = t1;
      }
    }
    __syncthreads();
    h16 afr[2], bfr[4];
#pragma unroll
    for (int mi = 0; mi < 2; ++mi) {
      const _Float16* p = &As[(wr * 32 + mi * 16 + r) * 40];
      afr[mi] = mk16(*(const h8*)(p + 16 * kb), *(const h8*)(p + 16 * kb + 8));
    }
#pragma unroll
    for (int ni = 0; ni < 4; ++ni) {
      const _Float16* p = &Bs[(wc * 64 + ni * 16 + r) * 40 + 16 * kb];
      bfr[ni] = mk16(*(const h8*)p, *(const h8*)(p + 8));
    }
#pragma unroll
    for (int mi = 0; mi < 2; ++mi)
#pragma unroll
      for (int ni = 0; ni < 4; ++ni) acc[mi][ni] = WMMA(afr[mi], bfr[ni], acc[mi][ni]);
    __syncthreads();
  }
#pragma unroll
  for (int ni = 0; ni < 4; ++ni) {
    const int col = n0 + wc * 64 + ni * 16 + r;
    const float bias = bqkv[col];
    const int s = col >> 10, cc = col & (C_ - 1);
    const int hh = cc >> 6, d = cc & (HD - 1);
    _Float16* dst = (s == 0) ? Qh : ((s == 1) ? Kh : Vh);
#pragma unroll
    for (int mi = 0; mi < 2; ++mi) {
#pragma unroll
      for (int v = 0; v < 8; ++v) {
        const int row = m0 + wr * 32 + mi * 16 + kb * 8 + v;
        const int bb = row >> 11, tt = row & (T_ - 1);
        dst[(size_t)(((bb * H_ + hh) * T_ + tt) * HD + d)] =
            (_Float16)(acc[mi][ni][v] + bias);
      }
    }
  }
}

// =====================================================================
// Kernel 2: flash attention. 1 block = (b,h, 64-row q block); 4 waves.
// K tile staged by the Tensor Data Mover (TDM) when available.
// =====================================================================
__global__ void __launch_bounds__(128) flash_attn_kernel(
    const _Float16* __restrict__ Qh, const _Float16* __restrict__ Kh,
    const _Float16* __restrict__ Vh, const int* __restrict__ mask,
    _Float16* __restrict__ Yh) {
  __shared__ _Float16 Ks[64 * 88];     // [kv][d] stride 88 (conflict-free b128 gathers)
  __shared__ _Float16 Vs[64 * 88];     // [d][kv] (transposed)
  __shared__ _Float16 Ps[4][16 * 88];  // per-wave P tile [q][kv]
  const int tid = threadIdx.x;
  const int lane = tid & 31, wv = tid >> 5;
  const int r = lane & 15, kb = lane >> 4;
  const int idx = blockIdx.x;
  const int qblk = idx & 31, hh = (idx >> 5) & 15, bb = idx >> 9;
  const int q0 = qblk * 64;
  const int qw0 = q0 + wv * 16;
  const size_t headoff = (size_t)(bb * H_ + hh) * T_ * HD;

  const _Float16* Qb = Qh + headoff + (size_t)qw0 * HD;
  h16 aQ[2];
#pragma unroll
  for (int kc = 0; kc < 2; ++kc) {
    const _Float16* p = Qb + r * HD + kc * 32;
    aQ[kc] = mk16(*(const h8*)(p + 16 * kb), *(const h8*)(p + 16 * kb + 8));
  }

  float mst[8], lst[8];
  f8 oacc[4];
#pragma unroll
  for (int v = 0; v < 8; ++v) { mst[v] = -1e30f; lst[v] = 0.f; }
#pragma unroll
  for (int n = 0; n < 4; ++n) oacc[n] = fzero8();
  const float scale = 0.125f;  // 1/sqrt(64)

  for (int kv0 = 0; kv0 < q0 + 64; kv0 += 64) {
    const _Float16* Kb = Kh + headoff + (size_t)kv0 * HD;
    const _Float16* Vb = Vh + headoff + (size_t)kv0 * HD;

#if USE_TDM
    // ---- K tile via Tensor Data Mover: 64x64 f16 tile, LDS pad to stride 88
    if (wv == 0) {
      const unsigned long long ga = (unsigned long long)(uintptr_t)Kb;
      const unsigned ldsoff = (unsigned)(uintptr_t)&Ks[0];  // low 32 bits = LDS offset
      u32x4 g0;
      g0[0] = 1u;                                    // count=1, user-mode descriptor
      g0[1] = ldsoff;                                // lds_addr (bytes)
      g0[2] = (unsigned)ga;                          // global_addr[31:0]
      g0[3] = (unsigned)(ga >> 32) | (2u << 30);     // global_addr[56:32] | type=2
      i32x8 g1;
      g1[0] = (1 << 16) | (1 << 20) | (4 << 22) | (11 << 25);
      //       data_size=2B | pad_en | pad_interval=32DW | pad_amount=12DW
      g1[1] = 64 << 16;    // tensor_dim0 = 64 (d)
      g1[2] = 2048 << 16;  // tensor_dim1 = 2048 (kv rows)
      g1[3] = 64 << 16;    // tile_dim0 = 64
      g1[4] = 64;          // tile_dim1 = 64, tile_dim2 = 0
      g1[5] = 64;          // tensor_dim0_stride = 64
      g1[6] = 0;
      g1[7] = 0;
      i32x4 g2 = {0, 0, 0, 0}, g3 = {0, 0, 0, 0};
#if __clang_major__ >= 23
      i32x8 g4 = {0, 0, 0, 0, 0, 0, 0, 0};
      __builtin_amdgcn_tensor_load_to_lds(g0, g1, g2, g3, g4, 0);
#else
      __builtin_amdgcn_tensor_load_to_lds(g0, g1, g2, g3, 0);
#endif
    }
#else
    {  // ---- manual K staging
      const int rr = tid >> 1, cc0 = (tid & 1) * 32;
#pragma unroll
      for (int j = 0; j < 4; ++j) {
        h8 tk = *(const h8*)(Kb + rr * HD + cc0 + 8 * j);
        *(h8*)&Ks[rr * 88 + cc0 + 8 * j] = tk;
      }
    }
#endif
    {  // ---- V staged transposed, packed-pair b32 stores
      const int rv = (tid >> 2) * 2;   // 0,2,...,62
      const int cv = (tid & 3) * 16;   // 0,16,32,48
      h8 v0a = *(const h8*)(Vb + rv * HD + cv);
      h8 v0b = *(const h8*)(Vb + rv * HD + cv + 8);
      h8 v1a = *(const h8*)(Vb + (rv + 1) * HD + cv);
      h8 v1b = *(const h8*)(Vb + (rv + 1) * HD + cv + 8);
#pragma unroll
      for (int i = 0; i < 8; ++i) {
        h2 ta = {v0a[i], v1a[i]};
        h2 tb = {v0b[i], v1b[i]};
        *(h2*)&Vs[(cv + i) * 88 + rv] = ta;
        *(h2*)&Vs[(cv + 8 + i) * 88 + rv] = tb;
      }
    }
#if USE_TDM
    if (wv == 0) __builtin_amdgcn_s_wait_tensorcnt(0);
#endif
    __syncthreads();

    // ---- S = (Q K^T) * scale, masked ----
    f8 sf[4];
#pragma unroll
    for (int n = 0; n < 4; ++n) {
      const _Float16* p0 = &Ks[(n * 16 + r) * 88 + 16 * kb];
      const _Float16* p1 = &Ks[(n * 16 + r) * 88 + 32 + 16 * kb];
      h16 b0 = mk16(*(const h8*)p0, *(const h8*)(p0 + 8));
      h16 b1 = mk16(*(const h8*)p1, *(const h8*)(p1 + 8));
      f8 z = fzero8();
      z = WMMA(aQ[0], b0, z);
      z = WMMA(aQ[1], b1, z);
      sf[n] = z;
    }
    const int kvbase = kv0 + r;
    int pm[4];
#pragma unroll
    for (int n = 0; n < 4; ++n) pm[n] = mask[bb * T_ + kvbase + n * 16];

    // ---- online softmax per row ----
#pragma unroll
    for (int v = 0; v < 8; ++v) {
      const int qpos = qw0 + kb * 8 + v;
      float sv[4];
      float rowmax = -1e30f;
#pragma unroll
      for (int n = 0; n < 4; ++n) {
        const int kvpos = kvbase + n * 16;
        float s = sf[n][v] * scale;
        if (kvpos > qpos || pm[n] == 0) s = -1e30f;
        sv[n] = s;
        rowmax = fmaxf(rowmax, s);
      }
#pragma unroll
      for (int off = 8; off > 0; off >>= 1)
        rowmax = fmaxf(rowmax, __shfl_xor(rowmax, off, 16));
      const float mnew = fmaxf(mst[v], rowmax);
      const float alpha = __expf(mst[v] - mnew);
      mst[v] = mnew;
      float rs = 0.f;
#pragma unroll
      for (int n = 0; n < 4; ++n) {
        const float p = __expf(sv[n] - mnew);
        rs += p;
        Ps[wv][(kb * 8 + v) * 88 + n * 16 + r] = (_Float16)p;
      }
#pragma unroll
      for (int off = 8; off > 0; off >>= 1) rs += __shfl_xor(rs, off, 16);
      lst[v] = lst[v] * alpha + rs;
#pragma unroll
      for (int n = 0; n < 4; ++n) oacc[n][v] *= alpha;
    }

    // ---- O += P @ V ----
    h16 aP[2];
#pragma unroll
    for (int kc = 0; kc < 2; ++kc) {
      const _Float16* p = &Ps[wv][r * 88 + kc * 32];
      aP[kc] = mk16(*(const h8*)(p + 16 * kb), *(const h8*)(p + 16 * kb + 8));
    }
#pragma unroll
    for (int n = 0; n < 4; ++n) {
#pragma unroll
      for (int kc = 0; kc < 2; ++kc) {
        const _Float16* p = &Vs[(n * 16 + r) * 88 + kc * 32 + 16 * kb];
        h16 bv = mk16(*(const h8*)p, *(const h8*)(p + 8));
        oacc[n] = WMMA(aP[kc], bv, oacc[n]);
      }
    }
    __syncthreads();
  }

#pragma unroll
  for (int v = 0; v < 8; ++v) {
    const float inv = 1.0f / lst[v];
    const int row = qw0 + kb * 8 + v;
    const size_t base = ((size_t)(bb * T_ + row)) * C_ + hh * HD;
#pragma unroll
    for (int n = 0; n < 4; ++n)
      Yh[base + n * 16 + r] = (_Float16)(oacc[n][v] * inv);
  }
}

// =====================================================================
// Kernel 3: out = y @ W_proj + b_proj (f32 out). M=4096,N=1024,K=1024.
// A tile staged with async global->LDS when available.
// =====================================================================
__global__ void __launch_bounds__(256) proj_gemm_kernel(
    const _Float16* __restrict__ Yh, const float* __restrict__ Wp,
    const float* __restrict__ bp, float* __restrict__ out) {
  __shared__ _Float16 As[128 * 40];
  __shared__ _Float16 Bs[128 * 40];
  const int tid = threadIdx.x;
  const int lane = tid & 31, wv = tid >> 5;
  const int wr = wv >> 1, wc = wv & 1;
  const int r = lane & 15, kb = lane >> 4;
  const int m0 = blockIdx.y * 128, n0 = blockIdx.x * 128;

  f8 acc[2][4];
#pragma unroll
  for (int mi = 0; mi < 2; ++mi)
#pragma unroll
    for (int ni = 0; ni < 4; ++ni) acc[mi][ni] = fzero8();

  for (int k0 = 0; k0 < C_; k0 += 32) {
    {  // stage A (f16 copy): async global->LDS path
      const int m = tid >> 1, kk = (tid & 1) * 16;
      const _Float16* src = Yh + (size_t)(m0 + m) * C_ + k0 + kk;
      if (k0 + 32 < C_) __builtin_prefetch(src + 32, 0, 1);
#if USE_ASYNC
      i32x4 __attribute__((address_space(1)))* g0p =
          (i32x4 __attribute__((address_space(1)))*)(uintptr_t)src;
      i32x4 __attribute__((address_space(1)))* g1p =
          (i32x4 __attribute__((address_space(1)))*)(uintptr_t)(src + 8);
      i32x4 __attribute__((address_space(3)))* l0p =
          (i32x4 __attribute__((address_space(3)))*)(uintptr_t)(unsigned)(uintptr_t)&As[m * 40 + kk];
      i32x4 __attribute__((address_space(3)))* l1p =
          (i32x4 __attribute__((address_space(3)))*)(uintptr_t)(unsigned)(uintptr_t)&As[m * 40 + kk + 8];
      __builtin_amdgcn_global_load_async_to_lds_b128(g0p, l0p, 0, 0);
      __builtin_amdgcn_global_load_async_to_lds_b128(g1p, l1p, 0, 0);
#else
      *(h8*)&As[m * 40 + kk] = *(const h8*)src;
      *(h8*)&As[m * 40 + kk + 8] = *(const h8*)(src + 8);
#endif
    }
    {  // stage B transposed, packed-pair b32 stores
      const int kkp = (tid >> 4) * 2;
      const int nb = (tid & 15) * 8;
      const float* s0 = Wp + (size_t)(k0 + kkp) * C_ + n0 + nb;
      const float* s1 = s0 + C_;
      if (k0 + 32 < C_) __builtin_prefetch(s0 + (size_t)32 * C_, 0, 1);
      f4 a0 = *(const f4*)(s0), a1 = *(const f4*)(s0 + 4);
      f4 b0 = *(const f4*)(s1), b1 = *(const f4*)(s1 + 4);
#pragma unroll
      for (int i = 0; i < 4; ++i) {
        h2 t0 = {(_Float16)a0[i], (_Float16)b0[i]};
        h2 t1 = {(_Float16)a1[i], (_Float16)b1[i]};
        *(h2*)&Bs[(nb + i) * 40 + kkp] = t0;
        *(h2*)&Bs[(nb + 4 + i) * 40 + kkp] = t1;
      }
    }
#if USE_ASYNC
    wait_async0();
#endif
    __syncthreads();
    h16 afr[2], bfr[4];
#pragma unroll
    for (int mi = 0; mi < 2; ++mi) {
      const _Float16* p = &As[(wr * 32 + mi * 16 + r) * 40];
      afr[mi] = mk16(*(const h8*)(p + 16 * kb), *(const h8*)(p + 16 * kb + 8));
    }
#pragma unroll
    for (int ni = 0; ni < 4; ++ni) {
      const _Float16* p = &Bs[(wc * 64 + ni * 16 + r) * 40 + 16 * kb];
      bfr[ni] = mk16(*(const h8*)p, *(const h8*)(p + 8));
    }
#pragma unroll
    for (int mi = 0; mi < 2; ++mi)
#pragma unroll
      for (int ni = 0; ni < 4; ++ni) acc[mi][ni] = WMMA(afr[mi], bfr[ni], acc[mi][ni]);
    __syncthreads();
  }
#pragma unroll
  for (int ni = 0; ni < 4; ++ni) {
    const int col = n0 + wc * 64 + ni * 16 + r;
    const float bias = bp[col];
#pragma unroll
    for (int mi = 0; mi < 2; ++mi) {
#pragma unroll
      for (int v = 0; v < 8; ++v) {
        const int row = m0 + wr * 32 + mi * 16 + kb * 8 + v;
        out[(size_t)row * C_ + col] = acc[mi][ni][v] + bias;
      }
    }
  }
}

// =====================================================================
extern "C" void kernel_launch(void* const* d_in, const int* in_sizes, int n_in,
                              void* d_out, int out_size, void* d_ws, size_t ws_size,
                              hipStream_t stream) {
  (void)in_sizes; (void)n_in; (void)out_size; (void)ws_size;
  const float* x    = (const float*)d_in[0];
  const int*   msk  = (const int*)d_in[1];
  const float* Wqkv = (const float*)d_in[2];
  const float* bqkv = (const float*)d_in[3];
  const float* Wp   = (const float*)d_in[4];
  const float* bp   = (const float*)d_in[5];
  float* out = (float*)d_out;

  const size_t headElems = (size_t)B_ * H_ * T_ * HD;
  _Float16* Qh = (_Float16*)d_ws;
  _Float16* Kh = Qh + headElems;
  _Float16* Vh = Kh + headElems;
  _Float16* Yh = Vh + headElems;

  qkv_gemm_kernel<<<dim3(N3C / 128, (B_ * T_) / 128), 256, 0, stream>>>(
      x, Wqkv, bqkv, Qh, Kh, Vh);
  flash_attn_kernel<<<dim3(B_ * H_ * (T_ / 64)), 128, 0, stream>>>(
      Qh, Kh, Vh, msk, Yh);
  proj_gemm_kernel<<<dim3(C_ / 128, (B_ * T_) / 128), 256, 0, stream>>>(
      Yh, Wp, bp, out);
}
